// LDConv_25262997635311
// MI455X (gfx1250) — matrix operand order repaired and checked by reading, not compiled, not run on hardware
//
#include <hip/hip_runtime.h>
#include <hip/hip_bf16.h>
#include <cmath>

typedef __bf16 bf16_t;
typedef __attribute__((ext_vector_type(16))) bf16_t bf16x16;
typedef __attribute__((ext_vector_type(8)))  bf16_t bf16x8;
typedef __attribute__((ext_vector_type(8)))  float  v8f;

#define B_   8
#define C_   128
#define H_   64
#define W_   64
#define HW_  4096
#define NP   9
#define OC   256
#define KK   1152    // C_*NP
#define MM   32768   // B_*HW_

__device__ __forceinline__ unsigned short f2bf(float f) {
  unsigned int u = __float_as_uint(f);
  u += 0x7fffu + ((u >> 16) & 1u);   // round-to-nearest-even
  return (unsigned short)(u >> 16);
}

// ---------------------------------------------------------------------------
// Kernel 1: 3x3 offset conv (128 -> 18 channels), pad 1, stride 1.
// ---------------------------------------------------------------------------
__global__ __launch_bounds__(256) void offset_conv_kernel(
    const float* __restrict__ x, const float* __restrict__ w,
    const float* __restrict__ bias, float* __restrict__ off) {
  int idx = blockIdx.x * 256 + threadIdx.x;   // B*18*HW = 589824
  int hw  = idx & (HW_ - 1);
  int t   = idx >> 12;                        // b*18 + ch
  int ch  = t % 18;
  int b   = t / 18;
  int y = hw >> 6, xx = hw & 63;
  float acc = bias[ch];
  const float* xb = x + (size_t)b * C_ * HW_;
  const float* wc = w + (size_t)ch * C_ * 9;
  for (int c = 0; c < C_; ++c) {
    const float* xc = xb + c * HW_;
    const float* wt = wc + c * 9;
#pragma unroll
    for (int dy = 0; dy < 3; ++dy) {
      int yy = y + dy - 1;
      if ((unsigned)yy >= (unsigned)H_) continue;
#pragma unroll
      for (int dx = 0; dx < 3; ++dx) {
        int xx2 = xx + dx - 1;
        if ((unsigned)xx2 >= (unsigned)W_) continue;
        acc = fmaf(xc[yy * W_ + xx2], wt[dy * 3 + dx], acc);
      }
    }
  }
  off[idx] = acc;
}

// ---------------------------------------------------------------------------
// Kernel 2: repack conv_w (o,c,n) -> bf16 Bmat [o][k] with k = n*128 + c
// ---------------------------------------------------------------------------
__global__ __launch_bounds__(256) void prep_b_kernel(
    const float* __restrict__ w, unsigned short* __restrict__ bm) {
  int idx = blockIdx.x * 256 + threadIdx.x;   // OC*KK = 294912
  int k = idx % KK;
  int o = idx / KK;
  int n = k >> 7;
  int c = k & 127;
  bm[idx] = f2bf(w[(size_t)o * KK + c * NP + n]);
}

// ---------------------------------------------------------------------------
// Kernel 3: bilinear gather -> x_off bf16 [m][k], k = n*128 + c.
// One thread per (m, n); loops channels in pairs, packed 32-bit stores.
// ---------------------------------------------------------------------------
__global__ __launch_bounds__(256) void gather_kernel(
    const float* __restrict__ x, const float* __restrict__ off,
    unsigned int* __restrict__ xoff32) {
  int idx = blockIdx.x * 256 + threadIdx.x;   // MM*NP = 294912
  int m = idx / NP, n = idx % NP;
  int b = m >> 12, hw = m & (HW_ - 1);
  int y = hw >> 6, xx = hw & 63;
  float offx = off[((size_t)(b * 18 + n)) * HW_ + hw];
  float offy = off[((size_t)(b * 18 + 9 + n)) * HW_ + hw];
  float px = (float)(y + n / 3) + offx;     // p_n grid for N=9: (n/3, n%3)
  float py = (float)(xx + n % 3) + offy;
  float fx = floorf(px), fy = floorf(py);
  float qltx = fminf(fmaxf(fx, 0.f), 63.f);
  float qlty = fminf(fmaxf(fy, 0.f), 63.f);
  float qrbx = fminf(fmaxf(fx + 1.f, 0.f), 63.f);
  float qrby = fminf(fmaxf(fy + 1.f, 0.f), 63.f);
  float pcx = fminf(fmaxf(px, 0.f), 63.f);
  float pcy = fminf(fmaxf(py, 0.f), 63.f);
  float glt = (1.f + qltx - pcx) * (1.f + qlty - pcy);
  float grb = (1.f - qrbx + pcx) * (1.f - qrby + pcy);
  float glb = (1.f + qltx - pcx) * (1.f - qrby + pcy);
  float grt = (1.f - qrbx + pcx) * (1.f + qlty - pcy);
  int ilt = (int)qltx * W_ + (int)qlty;
  int irb = (int)qrbx * W_ + (int)qrby;
  int ilb = (int)qltx * W_ + (int)qrby;
  int irt = (int)qrbx * W_ + (int)qlty;
  const float* xb = x + (size_t)b * C_ * HW_;
  unsigned int* dst = xoff32 + ((size_t)m * KK + n * C_) / 2;
  for (int c = 0; c < C_; c += 2) {
    const float* xc0 = xb + c * HW_;
    const float* xc1 = xc0 + HW_;
    float v0 = glt * xc0[ilt] + grb * xc0[irb] + glb * xc0[ilb] + grt * xc0[irt];
    float v1 = glt * xc1[ilt] + grb * xc1[irb] + glb * xc1[ilb] + grt * xc1[irt];
    dst[c >> 1] = (unsigned int)f2bf(v0) | ((unsigned int)f2bf(v1) << 16);
  }
}

// ---------------------------------------------------------------------------
// Kernel 4: WMMA bf16 GEMM  y[m][o] = sum_k A[m][k]*B[o][k], fused BN + SiLU.
// Block tile 128x128, 8 waves, wave tile 32(M)x64(N) = 2x4 of 16x16x32 WMMA.
// Staging via GLOBAL_LOAD_ASYNC_TO_LDS_B128 (ASYNCcnt), double-buffered LDS,
// stride 40 shorts (80 B) for conflict-free ds_load_b128 fragment reads.
// ---------------------------------------------------------------------------
#define TM 128
#define TN 128
#define TK 32
#define KT (KK / TK)   // 36
#define LSTR (TK + 8)  // 40 shorts = 80 bytes

__global__ __launch_bounds__(256) void gemm_bn_silu_kernel(
    const unsigned short* __restrict__ A, const unsigned short* __restrict__ Bm,
    const float* __restrict__ gamma, const float* __restrict__ beta,
    const float* __restrict__ mean, const float* __restrict__ var,
    float* __restrict__ out) {
  __shared__ __align__(16) unsigned short lA[2][TM][LSTR];
  __shared__ __align__(16) unsigned short lB[2][TN][LSTR];
  const int tid = threadIdx.x;
  const int m0 = blockIdx.x * TM;
  const int n0 = blockIdx.y * TN;
  const int wave = tid >> 5;
  const int lane = tid & 31;
  const int wm = wave & 3;    // M position of wave (0..3)
  const int wn = wave >> 2;   // N position of wave (0..1)
  const int half = lane >> 4;
  const int l16 = lane & 15;

  // Async copy of one 128x32 bf16 tile: each thread moves 2 x 16B straight
  // into LDS (no VGPR staging). Low 32 bits of the generic shared pointer are
  // the LDS byte offset (LDS aperture: ADDR[31:0] == LDS_ADDR).
  auto async_tile = [&](const unsigned short* __restrict__ G, int row0, int kt,
                        unsigned short (*L)[LSTR]) {
#pragma unroll
    for (int r = 0; r < 2; ++r) {
      int s = tid + 256 * r;
      int row = s >> 2, ch = s & 3;
      unsigned long long ga = (unsigned long long)(uintptr_t)(
          G + (size_t)(row0 + row) * KK + kt * TK + ch * 8);
      unsigned int la = (unsigned int)(uintptr_t)(&L[row][ch * 8]);
      asm volatile("global_load_async_to_lds_b128 %0, %1, off"
                   :: "v"(la), "v"(ga) : "memory");
    }
  };

  // A frag: lane<16 -> K {0..7,16..23}; lane>=16 -> K {8..15,24..31}
  auto fragA = [&](const unsigned short (*L)[LSTR], int mo) -> bf16x16 {
    const int row = mo + l16, kb = half * 8;
    bf16x8 lo = *reinterpret_cast<const bf16x8*>(&L[row][kb]);
    bf16x8 hi = *reinterpret_cast<const bf16x8*>(&L[row][kb + 16]);
    bf16x16 r;
#pragma unroll
    for (int i = 0; i < 8; ++i) { r[i] = lo[i]; r[8 + i] = hi[i]; }
    return r;
  };
  // B frag: lane<16 -> K {0..7,16..23}; lane>=16 -> K {8..15,24..31}
  auto fragB = [&](const unsigned short (*L)[LSTR], int no) -> bf16x16 {
    const int row = no + l16, kb = half * 8;
    bf16x8 lo = *reinterpret_cast<const bf16x8*>(&L[row][kb]);
    bf16x8 hi = *reinterpret_cast<const bf16x8*>(&L[row][kb + 16]);
    bf16x16 r;
#pragma unroll
    for (int i = 0; i < 8; ++i) { r[i] = lo[i]; r[8 + i] = hi[i]; }
    return r;
  };

  v8f acc[2][4];
#pragma unroll
  for (int i = 0; i < 2; ++i)
#pragma unroll
    for (int j = 0; j < 4; ++j)
#pragma unroll
      for (int r = 0; r < 8; ++r) acc[i][j][r] = 0.f;

  // Prologue: async-stage tile 0 into buffer 0, drain, barrier.
  async_tile(A, m0, 0, lA[0]);
  async_tile(Bm, n0, 0, lB[0]);
  asm volatile("s_wait_asynccnt 0x0" ::: "memory");
  __syncthreads();

  for (int kt = 0; kt < KT; ++kt) {
    const int cur = kt & 1;
    const bool more = (kt + 1) < KT;
    if (more) {
      // Buffer cur^1 was last READ before the previous barrier -> safe to fill.
      async_tile(A, m0, kt + 1, lA[cur ^ 1]);
      async_tile(Bm, n0, kt + 1, lB[cur ^ 1]);
    }
    bf16x16 af[2], bfm[4];
#pragma unroll
    for (int i = 0; i < 2; ++i) af[i] = fragA(lA[cur], wm * 32 + i * 16);
#pragma unroll
    for (int j = 0; j < 4; ++j) bfm[j] = fragB(lB[cur], wn * 64 + j * 16);
#pragma unroll
    for (int i = 0; i < 2; ++i)
#pragma unroll
      for (int j = 0; j < 4; ++j)
        acc[i][j] = __builtin_amdgcn_wmma_f32_16x16x32_bf16(
            false, af[i], false, bfm[j], (short)0, acc[i][j], false, false);
    // Drain this iteration's async copies (they overlapped the 8 WMMAs above),
    // then barrier so every wave sees the freshly filled buffer.
    asm volatile("s_wait_asynccnt 0x0" ::: "memory");
    __syncthreads();
  }

  // Epilogue: BN + SiLU, scatter to NCHW output.
  // C/D layout: VGPR r, lanes 0-15 -> M=r, N=lane; lanes 16-31 -> M=r+8, N=lane-16.
#pragma unroll
  for (int j = 0; j < 4; ++j) {
    const int nn = n0 + wn * 64 + j * 16 + l16;       // output channel
    const float sc = gamma[nn] * rsqrtf(var[nn] + 1e-5f);
    const float mu = mean[nn], bt = beta[nn];
#pragma unroll
    for (int i = 0; i < 2; ++i) {
      const int mbase = m0 + wm * 32 + i * 16 + half * 8;
#pragma unroll
      for (int r = 0; r < 8; ++r) {
        const int m = mbase + r;
        float yv = (acc[i][j][r] - mu) * sc + bt;
        yv = yv / (1.f + __expf(-yv));                 // SiLU
        const int bb = m >> 12, hw = m & (HW_ - 1);
        out[((size_t)(bb * OC + nn)) * HW_ + hw] = yv;
      }
    }
  }
}

// ---------------------------------------------------------------------------
extern "C" void kernel_launch(void* const* d_in, const int* in_sizes, int n_in,
                              void* d_out, int out_size, void* d_ws, size_t ws_size,
                              hipStream_t stream) {
  const float* x  = (const float*)d_in[0];
  const float* pw = (const float*)d_in[1];
  const float* pb = (const float*)d_in[2];
  const float* cw = (const float*)d_in[3];
  const float* g  = (const float*)d_in[4];
  const float* bt = (const float*)d_in[5];
  const float* mu = (const float*)d_in[6];
  const float* vr = (const float*)d_in[7];
  float* out = (float*)d_out;

  char* ws = (char*)d_ws;
  float* off           = (float*)ws;                                // 2,359,296 B
  unsigned short* xoff = (unsigned short*)(ws + 2359296);           // 75,497,472 B
  unsigned short* bm   = (unsigned short*)(ws + 2359296 + 75497472);// 589,824 B

  offset_conv_kernel<<<2304, 256, 0, stream>>>(x, pw, pb, off);
  prep_b_kernel<<<1152, 256, 0, stream>>>(cw, bm);
  gather_kernel<<<1152, 256, 0, stream>>>(x, off, (unsigned int*)xoff);
  gemm_bn_silu_kernel<<<dim3(MM / TM, OC / TN), 256, 0, stream>>>(
      xoff, bm, g, bt, mu, vr, out);
}
